// ParallelCNNDistribution_82532091560203
// MI455X (gfx1250) — matrix operand order repaired and checked by reading, not compiled, not run on hardware
//
#include <hip/hip_runtime.h>
#include <hip/hip_bf16.h>

// ---------------------------------------------------------------------------
// ParallelCNN log-prob, CDNA5 (gfx1250) implicit-GEMM WMMA implementation.
//
// All 21 autoregressive steps are independent (masks use ground-truth
// values), so one launch covers every (level, slice, channel) step.
// Per step: M = B*(res/2)^2 output pixels, N = 64 hidden, K = 171 -> 192.
// bf16 A/B, f32 accumulate via v_wmma_f32_16x16x32_bf16.
// Reveal predicate parity is constant per k-column -> per-k constant mask.
// Staging: batched loads (16 in flight), branchless selects, b64 LDS stores,
// two k-tiles per barrier round.
// ---------------------------------------------------------------------------

typedef __bf16 bf16_t;
typedef bf16_t v16bf __attribute__((ext_vector_type(16)));
typedef bf16_t v8bf  __attribute__((ext_vector_type(8)));
typedef float  v8f   __attribute__((ext_vector_type(8)));

#define TPB      256        // 8 waves of 32
#define MROWS    128        // M-tile rows per block (16 per wave)
#define APITCH   72         // bf16 elems per A row (144B: 16B-aligned, no bank conflicts)
#define BPITCH   200        // bf16 elems per B row in LDS (400B, 16B aligned)
#define NBLOCKS  11712      // 3*64 + 9*256 + 9*1024 tiles

union FragU  { v16bf v; v8bf h[2]; };
union Pack4U { bf16_t h[4]; unsigned long long u; };

__device__ __forceinline__ v16bf ld_frag(const bf16_t* lo, const bf16_t* hi) {
  FragU u;
  u.h[0] = *reinterpret_cast<const v8bf*>(lo);
  u.h[1] = *reinterpret_cast<const v8bf*>(hi);
  return u.v;
}

__global__ __launch_bounds__(TPB) void pcnn_step_gemm(
    const float* __restrict__ value,    // (32,3,128,128)
    const float* __restrict__ dparams,  // (32,16,128,128)
    const float* __restrict__ W1,       // (21,64,19,3,3) -> (21,64,171)
    const float* __restrict__ b1,       // (21,64)
    const float* __restrict__ W2,       // (21,1,64,1,1) -> (21,64)
    const float* __restrict__ b2,       // (21,1)
    float* __restrict__ partials,
    float* __restrict__ gout,
    int useAtomic) {
  __shared__ __align__(16) bf16_t sA[MROWS * APITCH];   // 18432 B
  __shared__ __align__(16) bf16_t sB[64 * BPITCH];      // 25600 B
  __shared__ float sb1[64];
  __shared__ float sw2[64];
  __shared__ float sRed[TPB];

  const int tid  = threadIdx.x;
  const int lane = tid & 31;
  const int wave = tid >> 5;

  // ---- step lookup from flat block id (scalar) ----
  const int bx = blockIdx.x;
  int st = 0, off = 0;
  for (st = 0; st < 21; ++st) {
    int rs = (st < 3) ? 32 : (64 << ((st - 3) / 9));
    int tiles = (rs * rs) >> 4;            // B*(rs/2)^2 / 128 = rs^2/16
    if (bx < off + tiles) break;
    off += tiles;
  }
  const int res = (st < 3) ? 32 : (64 << ((st - 3) / 9));
  int T, cch, idx, py, px;
  if (st < 3) { T = 0; cch = st; idx = st; py = 0; px = 0; }
  else {
    int u = st - 3, level = u / 9, s = (u % 9) / 3;
    cch = u % 3; T = s + 1;
    py = (s != 1); px = (s != 2);
    idx = 3 + level * 9 + s * 3 + cch;
  }
  const int r     = 128 / res;
  const int rsh   = (r == 4) ? 2 : ((r == 2) ? 1 : 0);
  const int ow    = res >> 1;
  const int oh    = res >> 1;
  const int mBase = (bx - off) * MROWS;

  // A block never crosses an image: rows/image = ow*oh >= 256 > 128.
  const int bb = mBase / (ow * oh);
  const float* valB = value   + (size_t)bb * 3  * 16384;
  const float* dpB  = dparams + (size_t)bb * 16 * 16384;

  // ---- stage weights: sB[n][k] = W1[idx][n][k], bf16, zero-padded ----
  {
    const int n = tid >> 2;                              // 4 threads per row
    const float* wrow = W1 + ((size_t)idx * 64 + n) * 171;
    for (int k = (tid & 3); k < BPITCH; k += 4) {
      float w = (k < 171) ? wrow[k] : 0.0f;
      sB[n * BPITCH + k] = (bf16_t)w;
    }
  }
  if (tid < 64) {
    sb1[tid] = b1[idx * 64 + tid];
    sw2[tid] = W2[idx * 64 + tid];
  }

  // ---- per-thread staging geometry: k-quad x 8 rows ----
  const int rg  = tid >> 4;         // row group 0..15 (rows rg + 16*i)
  const int kl4 = (tid & 15) * 4;   // klocal quad base within 64-wide stage
  int rowpack[8];
#pragma unroll
  for (int i = 0; i < 8; ++i) {
    int m  = mBase + rg + 16 * i;
    int ox = m % ow; int t2 = m / ow; int oy = t2 % oh;
    rowpack[i] = ((2 * oy + py) << 8) | (2 * ox + px);
  }

  __syncthreads();   // weights staged

  // ---- accumulators: 16(M) x 64(N) per wave, 4 N-tiles ----
  const v8f vzero = {0.f, 0.f, 0.f, 0.f, 0.f, 0.f, 0.f, 0.f};
  v8f acc[4];
#pragma unroll
  for (int i = 0; i < 4; ++i) acc[i] = vzero;

  const int col16 = lane & 15;
  const int hi    = lane >> 4;
  const bf16_t* pArow = sA + (wave * 16 + col16) * APITCH;

#pragma unroll 1
  for (int kt2 = 0; kt2 < 3; ++kt2) {       // two k-tiles per round
    // -- decode the thread's 4 k-columns once per round (branchless) --
    int dym1[4], dxm1[4], plOff[4], km[4];
    const float* bp[4];
#pragma unroll
    for (int j = 0; j < 4; ++j) {
      const int k   = kt2 * 64 + kl4 + j;
      const int ci  = (k * 57) >> 9;        // floor(k/9) for k < 192
      const int rem = k - ci * 9;
      const int dy  = (rem * 11) >> 5;      // floor(rem/3)
      const int dx  = rem - dy * 3;
      dym1[j] = dy - 1;
      dxm1[j] = dx - 1;
      const int isVal = (ci < 3) ? 1 : 0;
      const int ciC   = (ci < 18) ? ci : 18;
      plOff[j] = (isVal ? ciC : (ciC - 3)) << 14;     // *128*128
      bp[j]    = isVal ? valB : dpB;
      // reveal predicate: parity is constant per k (row parity fixed)
      const int ypar = (py + dy + 1) & 1;
      const int xpar = (px + dx + 1) & 1;
      const int slot = (0x1320 >> (((ypar << 1) | xpar) * 4)) & 0xF;
      const int Teff = T + ((ci < cch) ? 1 : 0);
      km[j] = ((k < 171) ? 1 : 0) & ((1 - isVal) | ((slot < Teff) ? 1 : 0));
    }

    // -- gather in two batches of 4 rows x 4 k (16 loads in flight) --
#pragma unroll
    for (int half = 0; half < 2; ++half) {
      float tv[16];
      int   sel[16];
#pragma unroll
      for (int i = 0; i < 4; ++i) {
        const int rp = rowpack[half * 4 + i];
        const int Yb = rp >> 8;
        const int Xb = rp & 0xFF;
#pragma unroll
        for (int j = 0; j < 4; ++j) {
          const int Y = Yb + dym1[j];
          const int X = Xb + dxm1[j];
          const int inb = (((unsigned)Y < (unsigned)res) ? 1 : 0) &
                          (((unsigned)X < (unsigned)res) ? 1 : 0) & km[j];
          int Yc = Y < 0 ? 0 : (Y > res - 1 ? res - 1 : Y);
          int Xc = X < 0 ? 0 : (X > res - 1 ? res - 1 : X);
          tv[i * 4 + j]  = bp[j][(Yc << (rsh + 7)) + (Xc << rsh) + plOff[j]];
          sel[i * 4 + j] = inb;
        }
      }
#pragma unroll
      for (int i = 0; i < 4; ++i) {
        Pack4U p;
#pragma unroll
        for (int j = 0; j < 4; ++j)
          p.h[j] = (bf16_t)(sel[i * 4 + j] ? tv[i * 4 + j] : 0.0f);
        const int row = rg + 16 * (half * 4 + i);
        *reinterpret_cast<unsigned long long*>(&sA[row * APITCH + kl4]) = p.u;
      }
    }
    __syncthreads();

    // -- WMMA: 2 A frags (16x32) per wave, 8 B frags (32x16) --
#pragma unroll
    for (int ktIn = 0; ktIn < 2; ++ktIn) {
      const int kt = kt2 * 2 + ktIn;
      const bf16_t* pa = pArow + ktIn * 32;
      const v16bf afrag = ld_frag(pa + hi * 8, pa + 16 + hi * 8);
#pragma unroll
      for (int nt = 0; nt < 4; ++nt) {
        const bf16_t* pB = sB + (nt * 16 + col16) * BPITCH + kt * 32;
        const v16bf bfrag = ld_frag(pB + hi * 8, pB + 16 + hi * 8);
        acc[nt] = __builtin_amdgcn_wmma_f32_16x16x32_bf16(
            false, afrag, false, bfrag, (short)0, acc[nt], false, false);
      }
    }
    __syncthreads();
  }

  // ---- epilogue: bias + ReLU + 1x1 conv + log-sigmoid terms ----
  float lpLocal = 0.0f;
  const float b2v = b2[idx];
#pragma unroll
  for (int v = 0; v < 8; ++v) {
    float t = 0.0f;
#pragma unroll
    for (int nt = 0; nt < 4; ++nt) {
      const int N = nt * 16 + col16;
      float h = acc[nt][v] + sb1[N];
      h = fmaxf(h, 0.0f);
      t += h * sw2[N];
    }
    // reduce across the 16 lanes holding the same output row
    t += __shfl_xor(t, 1);
    t += __shfl_xor(t, 2);
    t += __shfl_xor(t, 4);
    t += __shfl_xor(t, 8);
    if (col16 == 0) {
      const float z = t + b2v;
      const int m2  = mBase + wave * 16 + v + 8 * hi;
      const int ox2 = m2 % ow; int t3 = m2 / ow;
      const int oy2 = t3 % oh; const int bb2 = t3 / oh;
      const int Y2 = (2 * oy2 + py) << rsh;
      const int X2 = (2 * ox2 + px) << rsh;
      const float x = value[(((size_t)bb2 * 3 + cch) << 14) + (Y2 << 7) + X2];
      // x*logsig(z) + (1-x)*logsig(-z) = -(x*sp(-z) + (1-x)*sp(z))
      const float L   = log1pf(__expf(-fabsf(z)));
      const float spp = fmaxf(z, 0.0f) + L;   // softplus(z)
      const float spn = fmaxf(-z, 0.0f) + L;  // softplus(-z)
      lpLocal -= x * spn + (1.0f - x) * spp;
    }
  }

  // ---- deterministic block reduction ----
  sRed[tid] = lpLocal;
  __syncthreads();
  for (int sr = TPB / 2; sr > 0; sr >>= 1) {
    if (tid < sr) sRed[tid] += sRed[tid + sr];
    __syncthreads();
  }
  if (tid == 0) {
    if (useAtomic) atomicAdd(gout, sRed[0]);
    else           partials[blockIdx.x] = sRed[0];
  }
}

__global__ __launch_bounds__(TPB) void pcnn_reduce(
    const float* __restrict__ partials, int n, float* __restrict__ out) {
  __shared__ float s[TPB];
  float a = 0.0f;
  for (int i = threadIdx.x; i < n; i += TPB) a += partials[i];
  s[threadIdx.x] = a;
  __syncthreads();
  for (int k = TPB / 2; k > 0; k >>= 1) {
    if (threadIdx.x < k) s[threadIdx.x] += s[threadIdx.x + k];
    __syncthreads();
  }
  if (threadIdx.x == 0) out[0] = s[0];
}

extern "C" void kernel_launch(void* const* d_in, const int* in_sizes, int n_in,
                              void* d_out, int out_size, void* d_ws, size_t ws_size,
                              hipStream_t stream) {
  const float* value   = (const float*)d_in[0];
  const float* dparams = (const float*)d_in[1];
  const float* W1      = (const float*)d_in[2];
  const float* b1      = (const float*)d_in[3];
  const float* W2      = (const float*)d_in[4];
  const float* b2      = (const float*)d_in[5];
  float* out = (float*)d_out;

  const bool haveWs = (ws_size >= (size_t)NBLOCKS * sizeof(float));
  float* partials = haveWs ? (float*)d_ws : out;

  if (!haveWs) hipMemsetAsync(out, 0, sizeof(float), stream);

  pcnn_step_gemm<<<NBLOCKS, TPB, 0, stream>>>(
      value, dparams, W1, b1, W2, b2, partials, out, haveWs ? 0 : 1);

  if (haveWs)
    pcnn_reduce<<<1, TPB, 0, stream>>>(partials, NBLOCKS, out);
}
